// AtomMotifAttention_13529146982755
// MI455X (gfx1250) — matrix operand order repaired and checked by reading, not compiled, not run on hardware
//
#include <hip/hip_runtime.h>
#include <hip/hip_bf16.h>

// ---------------------------------------------------------------------------
// AtomMotifAttention for MI455X (gfx1250), wave32, WMMA bf16.
//
// Shapes: N_ATOMS=12288, N_MOTIFS=1536, DIM=256, HEADS=4, HEAD_DIM=64, G=128.
// d_out = [ out (12288*256 f32) | attn (12288*4*1536 f32) ]  (tuple, flat).
//
// Roofline: ~23 GFLOP vs ~0.37 GB of HBM traffic after fusing attn@V into the
// softmax kernel (attn fp32 is written once, never re-read) -> bandwidth
// bound at 23.3 TB/s; bf16 WMMA gives ample compute headroom.
// Workspace usage: ~20.8 MB of bf16 staging.
// ---------------------------------------------------------------------------

#define N_ATOMS   12288
#define N_MOTIFS  1536
#define DIM       256
#define NHEADS    4
#define HEAD_DIM  64
#define SROW      1540   // 1536 + 4 pad floats: row stride mod 64 banks == 4

typedef __attribute__((ext_vector_type(16))) __bf16 v16bf;
typedef __attribute__((ext_vector_type(8)))  __bf16 v8bf;
typedef __attribute__((ext_vector_type(8)))  float  v8f;

// ---- WMMA fragment helpers (layouts per CDNA5 ISA §7.12.2) -----
//
// A (16x32, bf16): lane l(0..15) holds row M=l; lane l+16 holds same row.
// => per lane: elements [0..7] = K kb+hi*8+0..7 ; [8..15] = K kb+16+hi*8+0..7.
__device__ __forceinline__ v16bf make_afrag(const __bf16* __restrict__ rowptr,
                                            int kb, int hi) {
  v8bf lo = *(const v8bf*)(rowptr + kb + hi * 8);
  v8bf up = *(const v8bf*)(rowptr + kb + 16 + hi * 8);
  v16bf a;
#pragma unroll
  for (int i = 0; i < 8; ++i) { a[i] = lo[i]; a[8 + i] = up[i]; }
  return a;
}

// B (32x16, bf16): lane l(0..15) holds column N=l; lane l+16 holds same column.
// => per lane: elements [0..7] = K kb+hi*8+0..7 ; [8..15] = K kb+16+hi*8+0..7
// (same K order as A).  With B^T (N x K) row-major: two 8-element chunks.
__device__ __forceinline__ v16bf make_bfrag(const __bf16* __restrict__ rowptr,
                                            int kb, int hi) {
  v8bf lo = *(const v8bf*)(rowptr + kb + hi * 8);
  v8bf up = *(const v8bf*)(rowptr + kb + 16 + hi * 8);
  v16bf b;
#pragma unroll
  for (int i = 0; i < 8; ++i) { b[i] = lo[i]; b[8 + i] = up[i]; }
  return b;
}

__device__ __forceinline__ v8f wmma_bf16(v16bf a, v16bf b, v8f c) {
  // D = A*B + C ; fp32 accumulate ; emits v_wmma_f32_16x16x32_bf16
  return __builtin_amdgcn_wmma_f32_16x16x32_bf16(
      /*neg_a=*/false, a, /*neg_b=*/false, b,
      /*c_mod=*/(short)0, c, /*reuse_a=*/false, /*reuse_b=*/false);
}

// ---------------------------------------------------------------------------
// Kernel 0: fp32 -> bf16 conversion (grid-stride).
// ---------------------------------------------------------------------------
__global__ void cvt_f32_bf16(const float* __restrict__ src,
                             __bf16* __restrict__ dst, int n) {
  for (int i = blockIdx.x * blockDim.x + threadIdx.x; i < n;
       i += gridDim.x * blockDim.x)
    dst[i] = (__bf16)src[i];
}

// ---------------------------------------------------------------------------
// Kernel 1: Y[M,256] = X[M,256] @ W^T[256,256] + bias.
// Block = 256 thr (8 waves); wave w owns 16x16 output tile (n0 = y*128+w*16).
// transpose_out: store Y[col*M + row] (used to build V^T [h*64+d][motif]).
// ---------------------------------------------------------------------------
__global__ void proj_kernel(const __bf16* __restrict__ X,
                            const __bf16* __restrict__ W,
                            const float* __restrict__ bias,
                            __bf16* __restrict__ Y, int M, int transpose_out) {
  const int lane = threadIdx.x & 31;
  const int wave = threadIdx.x >> 5;       // 0..7
  const int l = lane & 15, hi = lane >> 4; // lane-half
  const int m0 = blockIdx.x * 16;
  const int n0 = blockIdx.y * 128 + wave * 16;

  const __bf16* arow = X + (size_t)(m0 + l) * DIM;
  const __bf16* brow = W + (size_t)(n0 + l) * DIM;

  v8f acc = {};
#pragma unroll
  for (int kb = 0; kb < DIM; kb += 32)
    acc = wmma_bf16(make_afrag(arow, kb, hi), make_bfrag(brow, kb, hi), acc);

  const int col = n0 + l;
  const float bv = bias[col];
#pragma unroll
  for (int r = 0; r < 8; ++r) {
    const int row = m0 + hi * 8 + r;
    const float v = acc[r] + bv;
    if (transpose_out) Y[(size_t)col * M + row] = (__bf16)v;
    else               Y[(size_t)row * DIM + col] = (__bf16)v;
  }
}

// ---------------------------------------------------------------------------
// Kernel 2 (fused): masked scores + softmax + attn@V for one (16-atom, head).
// Block = 128 thr (4 waves). LDS holds the full 16 x 1536 fp32 score tile
// (~99 KB; WGP has 320 KB).  Phases:
//   1) WMMA QK^T over 96 motif tiles -> masked scores in LDS
//   2) cooperative row softmax; normalized probs written in-place to LDS and
//      streamed to d_out (coalesced fp32 attn output, write-once)
//   3) WMMA attn@V with A sourced from LDS (ds_load, fp32->bf16 in-register)
//      and B = Vt (L2-resident).  attn is never re-read from global.
// ---------------------------------------------------------------------------
__global__ void attention_kernel(const __bf16* __restrict__ Q,
                                 const __bf16* __restrict__ Km,
                                 const __bf16* __restrict__ Vt,
                                 const int* __restrict__ atom_batch,
                                 const int* __restrict__ motif_batch,
                                 float* __restrict__ attn,
                                 __bf16* __restrict__ ctx) {
  __shared__ float s[16][SROW];       // ~98.6 KB, bank-staggered rows
  __shared__ float red[16][8];
  __shared__ float rowstat[16];
  __shared__ int   ab[16];

  const int a0   = blockIdx.x * 16;
  const int h    = blockIdx.y;
  const int tid  = threadIdx.x;       // 0..127
  const int lane = tid & 31;
  const int wave = tid >> 5;          // 0..3
  const int l = lane & 15, hi = lane >> 4;

  if (tid < 16) ab[tid] = atom_batch[a0 + tid];
  __syncthreads();

  // ---- Phase 1: scores = (Q K^T) / 8, masked -------------------------------
  const __bf16* qrow = Q + (size_t)(a0 + l) * DIM + h * HEAD_DIM;
  const v16bf qa0 = make_afrag(qrow, 0, hi);
  const v16bf qa1 = make_afrag(qrow, 32, hi);

  for (int j = wave; j < N_MOTIFS / 16; j += 4) {
    const int n0 = j * 16;
    const __bf16* krow = Km + (size_t)(n0 + l) * DIM + h * HEAD_DIM;
    __builtin_prefetch(Km + (size_t)(n0 + 64 + l) * DIM + h * HEAD_DIM, 0, 0);
    v8f acc = {};
    acc = wmma_bf16(qa0, make_bfrag(krow, 0, hi), acc);
    acc = wmma_bf16(qa1, make_bfrag(krow, 32, hi), acc);
    const int mb = motif_batch[n0 + l];
#pragma unroll
    for (int r = 0; r < 8; ++r) {
      const int row = hi * 8 + r;
      const float v = acc[r] * 0.125f;   // 1/sqrt(64)
      s[row][n0 + l] = (ab[row] == mb) ? v : -1.0e9f;
    }
  }
  __syncthreads();

  // ---- Phase 2: row softmax (8 threads per row, 192 columns each) ----------
  const int row = tid >> 3;
  const int sub = tid & 7;
  const int c0  = sub * (N_MOTIFS / 8);

  float mx = -3.0e38f;
  for (int c = 0; c < N_MOTIFS / 8; ++c) mx = fmaxf(mx, s[row][c0 + c]);
  red[row][sub] = mx;
  __syncthreads();
  if (sub == 0) {
    float m2 = red[row][0];
#pragma unroll
    for (int i = 1; i < 8; ++i) m2 = fmaxf(m2, red[row][i]);
    rowstat[row] = m2;
  }
  __syncthreads();
  const float rmax = rowstat[row];

  float sum = 0.f;
  for (int c = 0; c < N_MOTIFS / 8; ++c) {
    const float e = __expf(s[row][c0 + c] - rmax);
    s[row][c0 + c] = e;
    sum += e;
  }
  __syncthreads();            // rowstat consumed by all before re-write
  red[row][sub] = sum;
  __syncthreads();
  if (sub == 0) {
    float t = 0.f;
#pragma unroll
    for (int i = 0; i < 8; ++i) t += red[row][i];
    rowstat[row] = 1.0f / t;
  }
  __syncthreads();
  const float rinv = rowstat[row];

  // Normalize in place and stream attn to global (write-once, coalesced).
  float* orow = attn + ((size_t)(a0 + row) * NHEADS + h) * N_MOTIFS;
  for (int c = 0; c < N_MOTIFS / 8; ++c) {
    const float p = s[row][c0 + c] * rinv;
    s[row][c0 + c] = p;
    orow[c0 + c] = p;
  }
  __syncthreads();

  // ---- Phase 3: ctx = attn @ V, A-fragments from LDS -----------------------
  const int d0 = wave * 16;                     // this wave's head-dim slice
  const __bf16* brow = Vt + (size_t)(h * HEAD_DIM + d0 + l) * N_MOTIFS;
  v8f acc = {};
  for (int kb = 0; kb < N_MOTIFS; kb += 32) {
    const float4 q0 = *(const float4*)(&s[l][kb + hi * 8]);
    const float4 q1 = *(const float4*)(&s[l][kb + hi * 8 + 4]);
    const float4 q2 = *(const float4*)(&s[l][kb + 16 + hi * 8]);
    const float4 q3 = *(const float4*)(&s[l][kb + 16 + hi * 8 + 4]);
    v16bf a;
    a[0]  = (__bf16)q0.x; a[1]  = (__bf16)q0.y; a[2]  = (__bf16)q0.z; a[3]  = (__bf16)q0.w;
    a[4]  = (__bf16)q1.x; a[5]  = (__bf16)q1.y; a[6]  = (__bf16)q1.z; a[7]  = (__bf16)q1.w;
    a[8]  = (__bf16)q2.x; a[9]  = (__bf16)q2.y; a[10] = (__bf16)q2.z; a[11] = (__bf16)q2.w;
    a[12] = (__bf16)q3.x; a[13] = (__bf16)q3.y; a[14] = (__bf16)q3.z; a[15] = (__bf16)q3.w;
    acc = wmma_bf16(a, make_bfrag(brow, kb, hi), acc);
  }
#pragma unroll
  for (int r = 0; r < 8; ++r) {
    const int arow_g = a0 + hi * 8 + r;
    ctx[(size_t)arow_g * DIM + h * HEAD_DIM + d0 + l] = (__bf16)acc[r];
  }
}

// ---------------------------------------------------------------------------
// Kernel 3: out = LN(ctx @ Wo^T + bo + atom_x) * g + b.
// Block = 512 thr (16 waves): waves cover 16 N-tiles; then wave w LayerNorms
// row w of the 16x256 LDS tile with __shfl_xor (wave32) reductions.
// ---------------------------------------------------------------------------
__global__ void outproj_ln_kernel(const __bf16* __restrict__ ctx,
                                  const __bf16* __restrict__ Wo,
                                  const float* __restrict__ bo,
                                  const float* __restrict__ xres,
                                  const float* __restrict__ ln_g,
                                  const float* __restrict__ ln_b,
                                  float* __restrict__ out) {
  __shared__ float t[16][DIM];   // 16 KB
  const int m0   = blockIdx.x * 16;
  const int lane = threadIdx.x & 31;
  const int wave = threadIdx.x >> 5;   // 0..15
  const int l = lane & 15, hi = lane >> 4;
  const int n0 = wave * 16;

  const __bf16* arow = ctx + (size_t)(m0 + l) * DIM;
  const __bf16* brow = Wo + (size_t)(n0 + l) * DIM;

  v8f acc = {};
#pragma unroll
  for (int kb = 0; kb < DIM; kb += 32)
    acc = wmma_bf16(make_afrag(arow, kb, hi), make_bfrag(brow, kb, hi), acc);

  const int col = n0 + l;
  const float bb = bo[col];
#pragma unroll
  for (int r = 0; r < 8; ++r) {
    const int row = hi * 8 + r;
    t[row][col] = acc[r] + bb + xres[(size_t)(m0 + row) * DIM + col];
  }
  __syncthreads();

  // LayerNorm: wave `wave` handles row `wave`; each lane owns 8 columns.
  const int row = wave;
  float sum = 0.f, sq = 0.f;
#pragma unroll
  for (int i = 0; i < 8; ++i) {
    const float v = t[row][lane * 8 + i];
    sum += v; sq += v * v;
  }
#pragma unroll
  for (int off = 16; off > 0; off >>= 1) {
    sum += __shfl_xor(sum, off, 32);
    sq  += __shfl_xor(sq,  off, 32);
  }
  const float mu  = sum * (1.0f / 256.0f);
  const float var = sq * (1.0f / 256.0f) - mu * mu;
  const float rs  = rsqrtf(var + 1e-5f);
#pragma unroll
  for (int i = 0; i < 8; ++i) {
    const int c = lane * 8 + i;
    out[(size_t)(m0 + row) * DIM + c] = (t[row][c] - mu) * rs * ln_g[c] + ln_b[c];
  }
}

// ---------------------------------------------------------------------------
// Host-side launch sequencing (graph-capture safe: only kernel launches).
// ---------------------------------------------------------------------------
extern "C" void kernel_launch(void* const* d_in, const int* in_sizes, int n_in,
                              void* d_out, int out_size, void* d_ws,
                              size_t ws_size, hipStream_t stream) {
  (void)in_sizes; (void)n_in; (void)out_size; (void)ws_size;

  const float* atom_x      = (const float*)d_in[0];
  const float* motif_x     = (const float*)d_in[1];
  const int*   atom_batch  = (const int*)d_in[2];
  const int*   motif_batch = (const int*)d_in[3];
  const float* Wq = (const float*)d_in[4];  const float* bq = (const float*)d_in[5];
  const float* Wk = (const float*)d_in[6];  const float* bk = (const float*)d_in[7];
  const float* Wv = (const float*)d_in[8];  const float* bv = (const float*)d_in[9];
  const float* Wo = (const float*)d_in[10]; const float* bo = (const float*)d_in[11];
  const float* ln_g = (const float*)d_in[12];
  const float* ln_b = (const float*)d_in[13];

  float* out  = (float*)d_out;                          // [12288, 256]
  float* attn = (float*)d_out + (size_t)N_ATOMS * DIM;  // [12288, 4, 1536]

  // Workspace layout (bf16 staging, ~20.8 MB total).
  char* ws = (char*)d_ws;
  __bf16* Xa   = (__bf16*)(ws);                         // 12288*256
  __bf16* Xm   = (__bf16*)(ws + 6291456);               //  1536*256
  __bf16* Wqb  = (__bf16*)(ws + 7077888);               //   256*256 x4
  __bf16* Wkb  = (__bf16*)(ws + 7208960);
  __bf16* Wvb  = (__bf16*)(ws + 7340032);
  __bf16* Wob  = (__bf16*)(ws + 7471104);
  __bf16* Qb   = (__bf16*)(ws + 7602176);               // 12288*256
  __bf16* Kb   = (__bf16*)(ws + 13893632);              //  1536*256
  __bf16* Vt   = (__bf16*)(ws + 14680064);              //   256*1536 (transposed)
  __bf16* ctx  = (__bf16*)(ws + 15466496);              // 12288*256

  // 0) fp32 -> bf16 staging.
  const int CT = 256;
  cvt_f32_bf16<<<(N_ATOMS * DIM + CT - 1) / CT, CT, 0, stream>>>(atom_x, Xa, N_ATOMS * DIM);
  cvt_f32_bf16<<<(N_MOTIFS * DIM + CT - 1) / CT, CT, 0, stream>>>(motif_x, Xm, N_MOTIFS * DIM);
  cvt_f32_bf16<<<(DIM * DIM + CT - 1) / CT, CT, 0, stream>>>(Wq, Wqb, DIM * DIM);
  cvt_f32_bf16<<<(DIM * DIM + CT - 1) / CT, CT, 0, stream>>>(Wk, Wkb, DIM * DIM);
  cvt_f32_bf16<<<(DIM * DIM + CT - 1) / CT, CT, 0, stream>>>(Wv, Wvb, DIM * DIM);
  cvt_f32_bf16<<<(DIM * DIM + CT - 1) / CT, CT, 0, stream>>>(Wo, Wob, DIM * DIM);

  // 1) Projections (WMMA bf16): Q, K row-major; V transposed per head-dim.
  proj_kernel<<<dim3(N_ATOMS / 16, 2), 256, 0, stream>>>(Xa, Wqb, bq, Qb, N_ATOMS, 0);
  proj_kernel<<<dim3(N_MOTIFS / 16, 2), 256, 0, stream>>>(Xm, Wkb, bk, Kb, N_MOTIFS, 0);
  proj_kernel<<<dim3(N_MOTIFS / 16, 2), 256, 0, stream>>>(Xm, Wvb, bv, Vt, N_MOTIFS, 1);

  // 2) Fused masked scores + softmax + attn@V (attn written once, LDS-fed GEMM).
  attention_kernel<<<dim3(N_ATOMS / 16, NHEADS), 128, 0, stream>>>(
      Qb, Kb, Vt, atom_batch, motif_batch, attn, ctx);

  // 3) out projection + bias + residual + LayerNorm.
  outproj_ln_kernel<<<N_ATOMS / 16, 512, 0, stream>>>(ctx, Wob, bo, atom_x,
                                                      ln_g, ln_b, out);
}
